// GruLayer_91036126806312
// MI455X (gfx1250) — matrix operand order repaired and checked by reading, not compiled, not run on hardware
//
#include <hip/hip_runtime.h>
#include <hip/hip_bf16.h>

// ---------------------------------------------------------------------------
// GRU-style recurrent layer for MI455X (gfx1250), wave32 + WMMA bf16.
//
// Phase 1 (parallel):  xproj[B*T,3072] = bf16(x) @ bf16([w_W_x | wx_W]) + bias
// Phase 2 (sequential, persistent kernel, 32 WGs, grid barrier):
//     WG's 192 KB slice of Wh_cat preloaded ONCE into LDS via
//     global_load_async_to_lds_b128 (ASYNCcnt), reused for all 512 steps.
//     S = bf16(h) @ bf16([w_W_h | wh_W])           (v_wmma_f32_16x16x32_bf16)
//     gh = sigmoid(xproj[:, 0:1024]  + S[:, 0:1024])
//     go = sigmoid(xproj[:,1024:2048]+ S[:,1024:2048])
//     m  = xproj[:,2048:3072] + gh * (S[:,2048:3072] + wh_b)
//     h  = go*h + (1-go)*tanh(m);  out[:,t,:] = h
// ---------------------------------------------------------------------------

typedef __attribute__((ext_vector_type(16))) __bf16 v16bf;
typedef __attribute__((ext_vector_type(8)))  float  v8f;

#define BB 64            // batch (M of recurrent GEMM)
#define TT 512
#define LL 1024
#define NN 3072          // fused N = 2*L (gates) + L (memory)
#define KK 1024
#define NBLK_REC 32      // persistent blocks (one per 32-col latent slice)
#define WPAD 1032        // LDS row pitch in elems (16B pad -> conflict-free banks)

__device__ __forceinline__ unsigned short f2bf(float f) {
    unsigned u = __float_as_uint(f);
    unsigned r = u + 0x7FFFu + ((u >> 16) & 1u);   // round-to-nearest-even
    return (unsigned short)(r >> 16);
}

__device__ __forceinline__ float sigmoidf(float x) {
    return 1.0f / (1.0f + __expf(-x));
}

// A fragment: row-major bf16, 16x32 tile; lane half h: contiguous K {kb..kb+15}
__device__ __forceinline__ v16bf load_fragA(const unsigned short* __restrict__ base,
                                            int pitch, int row, int kb) {
    union { v16bf v; uint4 q[2]; } u;
    const uint4* p = reinterpret_cast<const uint4*>(base + (size_t)row * pitch + kb);
    u.q[0] = p[0];        // K kb..kb+7
    u.q[1] = p[1];        // K kb+8..kb+15
    return u.v;
}

// B fragment: N-major (transposed) weights, 32x16 tile;
// lane half h: contiguous K {k0+16h .. k0+16h+15}
__device__ __forceinline__ v16bf load_fragB(const unsigned short* __restrict__ base,
                                            int pitch, int nrow, int k0h) {
    union { v16bf v; uint4 q[2]; } u;
    const uint4* p = reinterpret_cast<const uint4*>(base + (size_t)nrow * pitch + k0h);
    u.q[0] = p[0];
    u.q[1] = p[1];
    return u.v;
}

__device__ __forceinline__ v8f wmma_bf16(v16bf a, v16bf b, v8f c) {
    return __builtin_amdgcn_wmma_f32_16x16x32_bf16(false, a, false, b, (short)0, c,
                                                   false, false);
}

// ---------------------------------------------------------------------------
// Conversion kernels
// ---------------------------------------------------------------------------
__global__ void __launch_bounds__(256)
convx_kernel(const float* __restrict__ x, unsigned short* __restrict__ xb, int n) {
    int i = (blockIdx.x * 256 + threadIdx.x) * 8;
    if (i + 8 <= n) {
#pragma unroll
        for (int j = 0; j < 8; ++j) xb[i + j] = f2bf(x[i + j]);
    }
}

// Build transposed bf16 weight packs + fused x-side bias.
__global__ void __launch_bounds__(256)
buildw_kernel(const float* __restrict__ w_W, const float* __restrict__ w_b,
              const float* __restrict__ wx_W, const float* __restrict__ wx_b,
              const float* __restrict__ wh_W,
              unsigned short* __restrict__ WxT, unsigned short* __restrict__ WhT,
              float* __restrict__ biasx) {
    int idx = blockIdx.x * 256 + threadIdx.x;           // over 3072*1024
    if (idx >= NN * KK) return;
    int n = idx >> 10;          // 0..3071
    int k = idx & 1023;         // 0..1023
    float vx, vh;
    if (n < 2048) {
        vx = w_W[(size_t)k * 2048 + n];           // x-rows of w_W
        vh = w_W[(size_t)(1024 + k) * 2048 + n];  // h-rows of w_W
    } else {
        vx = wx_W[(size_t)k * 1024 + (n - 2048)];
        vh = wh_W[(size_t)k * 1024 + (n - 2048)];
    }
    WxT[(size_t)n * KK + k] = f2bf(vx);
    WhT[(size_t)n * KK + k] = f2bf(vh);
    if (k == 0) biasx[n] = (n < 2048) ? w_b[n] : wx_b[n - 2048];
}

// ---------------------------------------------------------------------------
// Phase 1: xproj = bf16(X)[32768,1024] @ WxT^T + biasx   (f32 out)
// grid: 512 row-blocks x 48 col-blocks of 64x64; 8 waves, each wave 2 tiles
// sharing one A fragment.
// ---------------------------------------------------------------------------
__global__ void __launch_bounds__(256)
xproj_kernel(const unsigned short* __restrict__ xb,
             const unsigned short* __restrict__ WxT,
             const float* __restrict__ biasx,
             float* __restrict__ xproj) {
    int bid  = blockIdx.x;
    int rb   = (bid & 511) * 64;
    int cb   = (bid >> 9) * 64;
    int lane = threadIdx.x & 31;
    int wv   = threadIdx.x >> 5;
    int half = lane >> 4;
    int r    = lane & 15;

    int m16  = (wv & 3) * 16;
    int n16  = (wv >> 2) * 16;        // tiles at n16 and n16+32
    int rowA = rb + m16 + r;

    v8f acc0 = {}, acc1 = {};
    for (int k0 = 0; k0 < KK; k0 += 32) {
        v16bf a  = load_fragA(xb, KK, rowA, k0 + half * 16);
        v16bf b0 = load_fragB(WxT, KK, cb + n16 + r,      k0 + half * 16);
        v16bf b1 = load_fragB(WxT, KK, cb + n16 + 32 + r, k0 + half * 16);
        acc0 = wmma_bf16(a, b0, acc0);
        acc1 = wmma_bf16(a, b1, acc1);
    }
    int c0 = cb + n16 + r, c1 = c0 + 32;
    float bv0 = biasx[c0], bv1 = biasx[c1];
#pragma unroll
    for (int q = 0; q < 8; ++q) {
        int m = rb + m16 + q + half * 8;
        xproj[(size_t)m * NN + c0] = acc0[q] + bv0;
        xproj[(size_t)m * NN + c1] = acc1[q] + bv1;
    }
}

// ---------------------------------------------------------------------------
// Phase 2: persistent recurrent kernel. 32 blocks x 256 threads.
// Grid barrier: sense-reversing atomic counter + s_sleep spin.
// ---------------------------------------------------------------------------
__device__ __forceinline__ void grid_barrier(unsigned* cnt, unsigned* gen) {
    __threadfence();
    __syncthreads();
    if (threadIdx.x == 0) {
        volatile unsigned* vgen = (volatile unsigned*)gen;
        unsigned g = *vgen;
        unsigned arrived = atomicAdd(cnt, 1u) + 1u;
        if (arrived == NBLK_REC) {
            atomicExch(cnt, 0u);
            __threadfence();
            atomicAdd(gen, 1u);
        } else {
            while (*vgen == g) { __builtin_amdgcn_s_sleep(2); }
        }
    }
    __syncthreads();
}

__global__ void __launch_bounds__(256)
recurrent_kernel(unsigned short* __restrict__ hbf,      // [2][64*1024] bf16, double buffer
                 float* __restrict__ hf32,              // [64*1024]
                 const unsigned short* __restrict__ WhT,// [3072][1024] bf16 (N-major)
                 const float* __restrict__ xproj,       // [32768][3072]
                 const float* __restrict__ whb,         // [1024]
                 const float* __restrict__ h0,          // [64*1024]
                 float* __restrict__ out,               // [64][512][1024]
                 unsigned* __restrict__ bar) {          // [0]=cnt [1]=gen
    extern __shared__ unsigned short wls[];             // [96][WPAD] bf16, ~193.5 KB

    // ---- one-time: async-copy this WG's Wh_cat slice (96 rows) into LDS ----
    // rows 0..31: hidden-gate cols, 32..63: output-gate, 64..95: memory
    for (int c = threadIdx.x; c < 96 * (KK / 8); c += 256) {
        int row96 = c >> 7, col = c & 127;                 // col in uint4 units
        int grow  = (row96 >> 5) * 1024 + blockIdx.x * 32 + (row96 & 31);
        const unsigned short* g = WhT + (size_t)grow * KK + col * 8;
        unsigned lds_off = (unsigned)(uintptr_t)(wls + (size_t)row96 * WPAD + col * 8);
        unsigned long long ga = (unsigned long long)(uintptr_t)g;
        asm volatile("global_load_async_to_lds_b128 %0, %1, off"
                     :: "v"(lds_off), "v"(ga) : "memory");
    }
    asm volatile("s_wait_asynccnt 0x0" ::: "memory");

    // ---- init h from h0 ----
    for (int i = blockIdx.x * 256 + threadIdx.x; i < BB * LL; i += NBLK_REC * 256) {
        float v = h0[i];
        hf32[i] = v;
        hbf[i]  = f2bf(v);          // parity-0 buffer
    }
    grid_barrier(bar, bar + 1);     // also covers LDS fill (__syncthreads inside)

    int lane = threadIdx.x & 31;
    int wv   = threadIdx.x >> 5;
    int half = lane >> 4;
    int r    = lane & 15;

    int m16 = (wv & 3) * 16;                       // M tile
    int jt  = (wv >> 2) * 16;                      // local N tile base (0 or 16)
    int j   = blockIdx.x * 32 + jt;                // global latent column tile

    for (int t = 0; t < TT; ++t) {
        int par = t & 1;
        const unsigned short* hb  = hbf + (size_t)par * (BB * LL);
        unsigned short*       hbn = hbf + (size_t)(1 - par) * (BB * LL);

        // ---- K-loop: A from L2 (hot 128 KB), B from persistent LDS ----
        v8f accH = {}, accO = {}, accM = {};
        int rowA = m16 + r;
        for (int k0 = 0; k0 < KK; k0 += 32) {
            v16bf a  = load_fragA(hb, KK, rowA, k0 + half * 16);
            int   kh = k0 + half * 16;
            v16bf bH = load_fragB(wls, WPAD, jt + r,      kh);   // ds_load_b128
            v16bf bO = load_fragB(wls, WPAD, 32 + jt + r, kh);
            v16bf bM = load_fragB(wls, WPAD, 64 + jt + r, kh);
            accH = wmma_bf16(a, bH, accH);
            accO = wmma_bf16(a, bO, accO);
            accM = wmma_bf16(a, bM, accM);
        }

        // ---- elementwise GRU update ----
        int   l     = j + r;
        float whb_l = whb[l];
#pragma unroll
        for (int q = 0; q < 8; ++q) {
            int m = m16 + q + half * 8;                      // batch row
            const float* xp = xproj + ((size_t)m * TT + t) * NN;
            float gh = sigmoidf(xp[l]        + accH[q]);
            float go = sigmoidf(xp[1024 + l] + accO[q]);
            float mm = xp[2048 + l] + gh * (accM[q] + whb_l);
            float ho = hf32[(size_t)m * LL + l];
            float hn = go * ho + (1.0f - go) * tanhf(mm);
            out[(size_t)m * (TT * LL) + (size_t)t * LL + l] = hn;
            hf32[(size_t)m * LL + l] = hn;
            hbn[(size_t)m * LL + l]  = f2bf(hn);
        }
        grid_barrier(bar, bar + 1);
    }
}

// ---------------------------------------------------------------------------
extern "C" void kernel_launch(void* const* d_in, const int* in_sizes, int n_in,
                              void* d_out, int out_size, void* d_ws, size_t ws_size,
                              hipStream_t stream) {
    const float* x    = (const float*)d_in[0];
    const float* h0   = (const float*)d_in[1];
    const float* w_W  = (const float*)d_in[2];
    const float* w_b  = (const float*)d_in[3];
    const float* wx_W = (const float*)d_in[4];
    const float* wx_b = (const float*)d_in[5];
    const float* wh_W = (const float*)d_in[6];
    const float* wh_b = (const float*)d_in[7];
    float* out = (float*)d_out;

    char* ws = (char*)d_ws;
    size_t OFF_XBF   = 0;                                   // 64 MB
    size_t OFF_WXT   = OFF_XBF   + (size_t)BB*TT*KK*2;      // 6 MB
    size_t OFF_WHT   = OFF_WXT   + (size_t)NN*KK*2;         // 6 MB
    size_t OFF_BIASX = OFF_WHT   + (size_t)NN*KK*2;         // 12 KB
    size_t OFF_HF32  = OFF_BIASX + (size_t)NN*4;            // 256 KB
    size_t OFF_HBF   = OFF_HF32  + (size_t)BB*LL*4;         // 256 KB
    size_t OFF_BAR   = OFF_HBF   + (size_t)2*BB*LL*2;       // 256 B
    size_t OFF_XPROJ = OFF_BAR   + 256;                     // 402 MB

    unsigned short* xb    = (unsigned short*)(ws + OFF_XBF);
    unsigned short* WxT   = (unsigned short*)(ws + OFF_WXT);
    unsigned short* WhT   = (unsigned short*)(ws + OFF_WHT);
    float*          biasx = (float*)(ws + OFF_BIASX);
    float*          hf32  = (float*)(ws + OFF_HF32);
    unsigned short* hbf   = (unsigned short*)(ws + OFF_HBF);
    unsigned*       bar   = (unsigned*)(ws + OFF_BAR);
    float*          xproj = (float*)(ws + OFF_XPROJ);

    int nx = BB * TT * KK;                                  // 33554432
    convx_kernel<<<nx / (256 * 8), 256, 0, stream>>>(x, xb, nx);
    buildw_kernel<<<(NN * KK) / 256, 256, 0, stream>>>(w_W, w_b, wx_W, wx_b, wh_W,
                                                       WxT, WhT, biasx);
    xproj_kernel<<<512 * 48, 256, 0, stream>>>(xb, WxT, biasx, xproj);
    hipMemsetAsync(bar, 0, 2 * sizeof(unsigned), stream);
    size_t lds_bytes = (size_t)96 * WPAD * sizeof(unsigned short);   // ~193.5 KB
    recurrent_kernel<<<NBLK_REC, 256, lds_bytes, stream>>>(hbf, hf32, WhT, xproj,
                                                           wh_b, h0, out, bar);
}
